// BertAttention_8272107012765
// MI455X (gfx1250) — matrix operand order repaired and checked by reading, not compiled, not run on hardware
//
#include <hip/hip_runtime.h>
#include <hip/hip_bf16.h>

// ---------------------------------------------------------------------------
// Problem constants (from reference)
// ---------------------------------------------------------------------------
constexpr int Bn  = 8;
constexpr int SQn = 512;
constexpr int SKn = 512;
constexpr int Hn  = 768;
constexpr int NHn = 12;
constexpr int DHn = 64;   // head dim
constexpr float SCALE = 0.125f;  // 1/sqrt(64)
constexpr float NEG_BIG = -3.0e38f;

typedef __attribute__((ext_vector_type(16))) __bf16 v16bf;
typedef __attribute__((ext_vector_type(8)))  float  v8f;

#define WMMA_BF16(a, b, c) \
  __builtin_amdgcn_wmma_f32_16x16x32_bf16(false, (a), false, (b), (short)0, (c), false, false)

// ---------------------------------------------------------------------------
// Fragment loaders (wave32; A/B use this file's contiguous k order, same for both)
// A 16x32 bf16: lane = {r=lane&15, half=lane>>4}; elems i = [r][16*half+i]
// B 32x16 bf16 (from row-major "W" with B[k][n] = W[n][k], i.e. NT GEMM):
//               lane column n=lane&15; elems i = W[n][16*half + i]
// C/D 16x16 f32: lane col n=lane&15; vgpr v -> row v + 8*(lane>>4)
// ---------------------------------------------------------------------------
__device__ __forceinline__ v16bf load_a_f32(const float* __restrict__ base, int ld, int lane) {
  const int r = lane & 15, hf = lane >> 4;
  const float* p = base + (size_t)r * ld + 16 * hf;
  v16bf f;
#pragma unroll
  for (int i = 0; i < 16; ++i) f[i] = (__bf16)p[i];
  return f;
}

__device__ __forceinline__ v16bf load_b_f32(const float* __restrict__ base, int ld, int lane) {
  const int n = lane & 15, hf = lane >> 4;
  const float* p = base + (size_t)n * ld + 16 * hf;
  v16bf f;
#pragma unroll
  for (int i = 0; i < 16; ++i) f[i] = (__bf16)p[i];
  return f;
}

__device__ __forceinline__ v16bf load_a_bf16(const __bf16* __restrict__ base, int ld, int lane) {
  const int r = lane & 15, hf = lane >> 4;
  const __bf16* p = base + (size_t)r * ld + 16 * hf;
  v16bf f;
#pragma unroll
  for (int i = 0; i < 16; ++i) f[i] = p[i];
  return f;
}

__device__ __forceinline__ v16bf load_b_bf16(const __bf16* __restrict__ base, int ld, int lane) {
  const int n = lane & 15, hf = lane >> 4;
  const __bf16* p = base + (size_t)n * ld + 16 * hf;
  v16bf f;
#pragma unroll
  for (int i = 0; i < 16; ++i) f[i] = p[i];
  return f;
}

// ---------------------------------------------------------------------------
// Kernel 0: one-pass f32 -> bf16 conversion (activations + weights), so the
// GEMM hot loops load bf16 directly (half the bytes, no cvts per k-step).
// n must be a multiple of 2048 (true for all tensors here).
// ---------------------------------------------------------------------------
__global__ __launch_bounds__(256) void cvt_kernel(const float* __restrict__ src,
                                                  __bf16* __restrict__ dst) {
  const size_t i = ((size_t)blockIdx.x * 256 + threadIdx.x) * 8;
#pragma unroll
  for (int k = 0; k < 8; ++k) dst[i + k] = (__bf16)src[i + k];
}

// ---------------------------------------------------------------------------
// Kernel 1: fused QKV projection (bf16 WMMA, f32 accumulate)
//   z==0: Q  = hidden @ Wq^T + bq            -> Qo[bh][s][d]     (bf16)
//   z==1: K' = context @ Wk^T + bk + pos_k   -> Ko[bh][s][d]     (bf16)
//   z==2: V  = context @ Wv^T + bv           -> Vo[bh][d][s]     (bf16, transposed)
// One wave computes a 32(M) x 64(N) strip (each B-frag feeds 2 WMMAs);
// K-loop over 768 in steps of 32: 12 b128 loads per 8 WMMAs.
// ---------------------------------------------------------------------------
__global__ __launch_bounds__(32) void qkv_kernel(
    const __bf16* __restrict__ Xh, const __bf16* __restrict__ Xc,
    const __bf16* __restrict__ Wqb, const __bf16* __restrict__ Wkb,
    const __bf16* __restrict__ Wvb,
    const float* __restrict__ bq, const float* __restrict__ bk,
    const float* __restrict__ bv,
    const float* __restrict__ ape,
    __bf16* __restrict__ Qo, __bf16* __restrict__ Ko, __bf16* __restrict__ Vo) {
  const int lane  = threadIdx.x;
  const int nbase = blockIdx.x * 64;
  const int mbase = blockIdx.y * 32;
  const int which = blockIdx.z;

  const __bf16* X = (which == 0) ? Xh : Xc;
  const __bf16* W = (which == 0) ? Wqb : (which == 1) ? Wkb : Wvb;
  const float* bias_ = (which == 0) ? bq : (which == 1) ? bk : bv;

  v8f z = {};
  v8f acc[2][4];
#pragma unroll
  for (int mt = 0; mt < 2; ++mt)
#pragma unroll
    for (int t = 0; t < 4; ++t) acc[mt][t] = z;

  for (int kk = 0; kk < Hn; kk += 32) {
    v16bf a0 = load_a_bf16(X + (size_t)mbase * Hn + kk, Hn, lane);
    v16bf a1 = load_a_bf16(X + (size_t)(mbase + 16) * Hn + kk, Hn, lane);
#pragma unroll
    for (int t = 0; t < 4; ++t) {
      v16bf b = load_b_bf16(W + (size_t)(nbase + 16 * t) * Hn + kk, Hn, lane);
      acc[0][t] = WMMA_BF16(a0, b, acc[0][t]);
      acc[1][t] = WMMA_BF16(a1, b, acc[1][t]);
    }
  }

  const int n = lane & 15, hf = lane >> 4;
#pragma unroll
  for (int mt = 0; mt < 2; ++mt) {
#pragma unroll
    for (int t = 0; t < 4; ++t) {
      const int c = nbase + 16 * t + n;     // global output column in [0,768)
      const int hh = c >> 6, d = c & 63;    // head, dim-in-head
      const float bias = bias_[c];
#pragma unroll
      for (int v = 0; v < 8; ++v) {
        const int m = mbase + 16 * mt + v + 8 * hf;  // global row in [0,4096)
        const int bb = m >> 9, s = m & 511;          // batch, seq
        const size_t bh = (size_t)bb * NHn + hh;
        float val = acc[mt][t][v] + bias;
        if (which == 0) {
          Qo[(bh * SQn + s) * DHn + d] = (__bf16)val;
        } else if (which == 1) {
          val += ape[(size_t)s * DHn + d];  // fold pos_k into K
          Ko[(bh * SKn + s) * DHn + d] = (__bf16)val;
        } else {
          Vo[(bh * DHn + d) * SKn + s] = (__bf16)val;  // transposed store
        }
      }
    }
  }
}

// ---------------------------------------------------------------------------
// Kernel 2: Pt = pos_k @ pos_q^T  (i.e. transposed positional scores,
// Pt[k][q] = sum_d ape[k][d]*ape[q][d]); computed in that orientation so its
// stores are coalesced AND the attention kernel reads 8 consecutive q-rows
// per lane contiguously (b128 instead of 8 scattered b32).
// ---------------------------------------------------------------------------
__global__ __launch_bounds__(32) void pos_kernel(const float* __restrict__ ape,
                                                 float* __restrict__ Pt) {
  const int lane = threadIdx.x;
  const int qb = blockIdx.x * 16;
  const int kb = blockIdx.y * 16;
  v8f acc = {};
#pragma unroll
  for (int kk = 0; kk < DHn; kk += 32) {
    v16bf a = load_a_f32(ape + (size_t)kb * DHn + kk, DHn, lane);  // M = key rows
    v16bf b = load_b_f32(ape + (size_t)qb * DHn + kk, DHn, lane);  // N = query rows
    acc = WMMA_BF16(a, b, acc);
  }
  const int n = lane & 15, hf = lane >> 4;
#pragma unroll
  for (int v = 0; v < 8; ++v)
    Pt[(size_t)(kb + v + 8 * hf) * SQn + qb + n] = acc[v];
}

// ---------------------------------------------------------------------------
// Kernel 3: flash attention per (b,h) per 16 query rows.
//   S = (Q @ K'^T + P) * scale + mask_bias ; online softmax ; O += exp(S) @ V
// One wave; key loop in steps of 32 (two 16-wide score tiles per step).
// exp(S) is bounced through LDS to convert C/D layout -> A-fragment layout.
// ---------------------------------------------------------------------------
__global__ __launch_bounds__(32) void attn_kernel(
    const __bf16* __restrict__ Q, const __bf16* __restrict__ K,
    const __bf16* __restrict__ Vt, const float* __restrict__ Pt,
    const int* __restrict__ amask, float* __restrict__ out) {
  __shared__ __bf16 plds[16 * 32];

  const int lane = threadIdx.x;
  const int n = lane & 15, hf = lane >> 4;
  const int bh = blockIdx.y, b = bh / NHn, h = bh % NHn;
  const int qbase = blockIdx.x * 16;

  const __bf16* Qh = Q  + ((size_t)bh * SQn + qbase) * DHn;  // [s][d]
  const __bf16* Kh = K  + (size_t)bh * SKn * DHn;            // [s][d]
  const __bf16* Vh = Vt + (size_t)bh * DHn * SKn;            // [d][s]
  const int* mk = amask + (size_t)b * SKn;

  // Q A-fragments for d = 0..31 and 32..63 (kept in registers all loop)
  const v16bf qa0 = load_a_bf16(Qh,      DHn, lane);
  const v16bf qa1 = load_a_bf16(Qh + 32, DHn, lane);

  v8f o0 = {}, o1 = {}, o2 = {}, o3 = {};
  float mrow[8], lrow[8];
#pragma unroll
  for (int v = 0; v < 8; ++v) { mrow[v] = NEG_BIG; lrow[v] = 0.0f; }

  for (int j = 0; j < SKn; j += 32) {
    // ---- scores: two 16x16 tiles covering key cols j..j+31 ----
    v8f s0 = {}, s1 = {};
    {
      v16bf kb0 = load_b_bf16(Kh + (size_t)j * DHn,        DHn, lane);
      v16bf kb1 = load_b_bf16(Kh + (size_t)(j + 16) * DHn, DHn, lane);
      s0 = WMMA_BF16(qa0, kb0, s0);
      s1 = WMMA_BF16(qa0, kb1, s1);
      kb0 = load_b_bf16(Kh + (size_t)j * DHn + 32,        DHn, lane);
      kb1 = load_b_bf16(Kh + (size_t)(j + 16) * DHn + 32, DHn, lane);
      s0 = WMMA_BF16(qa1, kb0, s0);
      s1 = WMMA_BF16(qa1, kb1, s1);
    }

    const int c0 = j + n, c1 = j + 16 + n;
    const float bias0 = mk[c0] ? 0.0f : NEG_BIG;
    const float bias1 = mk[c1] ? 0.0f : NEG_BIG;
    // transposed positional scores: 8 consecutive floats per lane per tile
    const float* pt0 = Pt + (size_t)c0 * SQn + qbase + 8 * hf;
    const float* pt1 = Pt + (size_t)c1 * SQn + qbase + 8 * hf;

    float cm[8];
#pragma unroll
    for (int v = 0; v < 8; ++v) {
      s0[v] = (s0[v] + pt0[v]) * SCALE + bias0;
      s1[v] = (s1[v] + pt1[v]) * SCALE + bias1;
      cm[v] = fmaxf(s0[v], s1[v]);
    }
    // row-max across the 16 lanes of each half (xor masks stay inside half)
#pragma unroll
    for (int msk = 1; msk < 16; msk <<= 1)
#pragma unroll
      for (int v = 0; v < 8; ++v) cm[v] = fmaxf(cm[v], __shfl_xor(cm[v], msk, 32));

    float alpha[8], rs[8];
#pragma unroll
    for (int v = 0; v < 8; ++v) {
      const float mn = fmaxf(mrow[v], cm[v]);
      alpha[v] = __expf(mrow[v] - mn);
      mrow[v] = mn;
      s0[v] = __expf(s0[v] - mn);
      s1[v] = __expf(s1[v] - mn);
      rs[v] = s0[v] + s1[v];
    }
#pragma unroll
    for (int msk = 1; msk < 16; msk <<= 1)
#pragma unroll
      for (int v = 0; v < 8; ++v) rs[v] += __shfl_xor(rs[v], msk, 32);
#pragma unroll
    for (int v = 0; v < 8; ++v) {
      lrow[v] = lrow[v] * alpha[v] + rs[v];
      o0[v] *= alpha[v]; o1[v] *= alpha[v]; o2[v] *= alpha[v]; o3[v] *= alpha[v];
    }

    // ---- C/D-layout -> A-layout via LDS (intra-wave, DS ops are in-order) --
#pragma unroll
    for (int v = 0; v < 8; ++v) {
      const int m = v + 8 * hf;
      plds[m * 32 + n]      = (__bf16)s0[v];
      plds[m * 32 + 16 + n] = (__bf16)s1[v];
    }
    asm volatile("" ::: "memory");
    v16bf pa;
    {
      const __bf16* pr = plds + (lane & 15) * 32;
#pragma unroll
      for (int i = 0; i < 16; ++i) pa[i] = pr[16 * hf + i];
    }
    asm volatile("" ::: "memory");

    // ---- O += exp(S) @ V  (B operand = rows of V^T, contiguous) ----
    o0 = WMMA_BF16(pa, load_b_bf16(Vh + (size_t) 0 * SKn + j, SKn, lane), o0);
    o1 = WMMA_BF16(pa, load_b_bf16(Vh + (size_t)16 * SKn + j, SKn, lane), o1);
    o2 = WMMA_BF16(pa, load_b_bf16(Vh + (size_t)32 * SKn + j, SKn, lane), o2);
    o3 = WMMA_BF16(pa, load_b_bf16(Vh + (size_t)48 * SKn + j, SKn, lane), o3);
  }

  // ---- epilogue: normalize and store fp32 output [b][s][h*64+d] ----
  v8f oo[4] = {o0, o1, o2, o3};
#pragma unroll
  for (int v = 0; v < 8; ++v) {
    const float inv = 1.0f / lrow[v];
    const size_t row = ((size_t)b * SQn + qbase + v + 8 * hf) * Hn + h * DHn;
#pragma unroll
    for (int t = 0; t < 4; ++t) out[row + 16 * t + n] = oo[t][v] * inv;
  }
}

// ---------------------------------------------------------------------------
// Host launcher
// d_in order: hidden, context, attention_mask(int), Wq, bq, Wk, bk, Wv, bv, ape
// ---------------------------------------------------------------------------
extern "C" void kernel_launch(void* const* d_in, const int* in_sizes, int n_in,
                              void* d_out, int out_size, void* d_ws, size_t ws_size,
                              hipStream_t stream) {
  const float* hidden  = (const float*)d_in[0];
  const float* context = (const float*)d_in[1];
  const int*   amask   = (const int*)  d_in[2];
  const float* Wq = (const float*)d_in[3];
  const float* bq = (const float*)d_in[4];
  const float* Wk = (const float*)d_in[5];
  const float* bk = (const float*)d_in[6];
  const float* Wv = (const float*)d_in[7];
  const float* bv = (const float*)d_in[8];
  const float* ape = (const float*)d_in[9];
  float* out = (float*)d_out;

  const size_t headElems = (size_t)Bn * NHn * SQn * DHn;  // 3,145,728
  const size_t xElems    = (size_t)Bn * SQn * Hn;         // 3,145,728
  const size_t wElems    = (size_t)Hn * Hn;               // 589,824

  __bf16* Qb  = (__bf16*)d_ws;
  __bf16* Kb  = Qb + headElems;
  __bf16* Vb  = Kb + headElems;
  float*  Ptb = (float*)(Vb + headElems);   // 512*512 f32 (1 MB)
  __bf16* Xhb = (__bf16*)(Ptb + (size_t)SKn * SQn);
  __bf16* Xcb = Xhb + xElems;
  __bf16* Wqb = Xcb + xElems;
  __bf16* Wkb = Wqb + wElems;
  __bf16* Wvb = Wkb + wElems;               // total ws ~36 MB

  // 0) one-pass bf16 conversion of activations and weights
  cvt_kernel<<<dim3((unsigned)(xElems / 2048)), 256, 0, stream>>>(hidden,  Xhb);
  cvt_kernel<<<dim3((unsigned)(xElems / 2048)), 256, 0, stream>>>(context, Xcb);
  cvt_kernel<<<dim3((unsigned)(wElems / 2048)), 256, 0, stream>>>(Wq, Wqb);
  cvt_kernel<<<dim3((unsigned)(wElems / 2048)), 256, 0, stream>>>(Wk, Wkb);
  cvt_kernel<<<dim3((unsigned)(wElems / 2048)), 256, 0, stream>>>(Wv, Wvb);

  // 1) fused QKV projection (z: 0=Q, 1=K+pos_k, 2=V transposed)
  dim3 g1(Hn / 64, (Bn * SQn) / 32, 3);
  qkv_kernel<<<g1, 32, 0, stream>>>(Xhb, Xcb, Wqb, Wkb, Wvb, bq, bk, bv,
                                    ape, Qb, Kb, Vb);
  // 2) transposed positional score matrix Pt = pos_k @ pos_q^T (once)
  pos_kernel<<<dim3(SQn / 16, SKn / 16), 32, 0, stream>>>(ape, Ptb);
  // 3) flash attention
  attn_kernel<<<dim3(SQn / 16, Bn * NHn), 32, 0, stream>>>(Qb, Kb, Vb, Ptb,
                                                           amask, out);
}
